// get_fingerprint_40587440947383
// MI455X (gfx1250) — matrix operand order repaired and checked by reading, not compiled, not run on hardware
//
#include <hip/hip_runtime.h>
#include <math.h>

typedef __attribute__((ext_vector_type(8)))  float  v8f;
typedef __attribute__((ext_vector_type(16))) __bf16 v16bf;
typedef __attribute__((ext_vector_type(8)))  __bf16 v8bf;
typedef __attribute__((ext_vector_type(4)))  int    v4i;

#define B_    16
#define CIN   256
#define COUT  256
#define H_    64
#define W_    64
#define FD    128
#define LAT   128
#define NW    589824              // Cout*Cin*9
#define STY   4608                // 2*Cin*9
#define LRELU 0.01f

// ---- gfx1250 async global->LDS path (guarded; fallback keeps file compiling) ----
#if defined(__has_builtin)
#if __has_builtin(__builtin_amdgcn_global_load_async_to_lds_b128)
#define HAVE_ASYNC_LDS 1
#endif
#if __has_builtin(__builtin_amdgcn_s_wait_asynccnt)
#define HAVE_WAIT_ASYNC 1
#endif
#endif

#define AS1 __attribute__((address_space(1)))
#define AS3 __attribute__((address_space(3)))

__device__ __forceinline__ void async_copy_b128(const void* g, void* l) {
#ifdef HAVE_ASYNC_LDS
    __builtin_amdgcn_global_load_async_to_lds_b128((AS1 v4i*)g, (AS3 v4i*)l, 0, 0);
#else
    *(v8bf*)l = *(const v8bf*)g;   // 16B synchronous fallback
#endif
}

__device__ __forceinline__ void wait_async_lds() {
#ifdef HAVE_ASYNC_LDS
#ifdef HAVE_WAIT_ASYNC
    __builtin_amdgcn_s_wait_asynccnt(0);
#else
    asm volatile("s_wait_asynccnt 0x0" ::: "memory");
#endif
#endif
}

__device__ __forceinline__ float lrelu(float x) { return x >= 0.0f ? x : LRELU * x; }

__device__ __forceinline__ void block_reduce2(float& s, float& s2, float* sh) {
    int tid = threadIdx.x;
    sh[tid] = s; sh[256 + tid] = s2;
    __syncthreads();
    for (int off = 128; off > 0; off >>= 1) {
        if (tid < off) { sh[tid] += sh[tid + off]; sh[256 + tid] += sh[256 + tid + off]; }
        __syncthreads();
    }
    s = sh[0]; s2 = sh[256];
}

// ---------- y fp32 [b][ci][h][w] -> bf16 [b][h][w][ci] ----------
__global__ __launch_bounds__(256) void k_convert_y(const float* __restrict__ y, __bf16* __restrict__ yt) {
    int g = blockIdx.x * 256 + threadIdx.x;           // 16*64*64*256 total
    int ci = g & 255;
    int w  = (g >> 8) & 63;
    int h  = (g >> 14) & 63;
    int b  = g >> 20;
    yt[g] = (__bf16)y[((b * CIN + ci) * H_ + h) * W_ + w];
}

// ---------- h = lrelu(finger @ style_w1^T)  [16,128] ----------
__global__ __launch_bounds__(256) void k_h(const float* __restrict__ finger, const float* __restrict__ w1,
                                           float* __restrict__ h) {
    int g = blockIdx.x * 256 + threadIdx.x;           // 2048
    int l = g & 127, b = g >> 7;
    float s = 0.f;
    for (int f = 0; f < FD; ++f) s += finger[b * FD + f] * w1[l * FD + f];
    h[g] = lrelu(s);
}

// ---------- style = h @ style_w2^T  [16,4608] ----------
__global__ __launch_bounds__(256) void k_style(const float* __restrict__ h, const float* __restrict__ w2,
                                               float* __restrict__ style) {
    int g = blockIdx.x * 256 + threadIdx.x;           // 16*4608
    int j = g % STY, b = g / STY;
    float s = 0.f;
    for (int l = 0; l < LAT; ++l) s += h[b * LAT + l] * w2[j * LAT + l];
    style[g] = s;
}

// ---------- bias path ----------
__global__ __launch_bounds__(256) void k_bias1(const float* __restrict__ finger, const float* __restrict__ bw1,
                                               float* __restrict__ t1) {
    int g = blockIdx.x * 256 + threadIdx.x;           // 16*256
    int o = g & 255, b = g >> 8;
    float s = 0.f;
    for (int f = 0; f < FD; ++f) s += finger[b * FD + f] * bw1[o * FD + f];
    t1[g] = lrelu(s);
}

__global__ __launch_bounds__(256) void k_bias2(const float* __restrict__ t1, const float* __restrict__ bw2,
                                               const float* __restrict__ f_bias, const float* __restrict__ b_mask,
                                               const float* __restrict__ ori_bias, float* __restrict__ nbias) {
    int g = blockIdx.x * 256 + threadIdx.x;           // 16*256
    int o = g & 255, b = g >> 8;
    float s = 0.f;
    for (int k = 0; k < COUT; ++k) s += t1[b * COUT + k] * bw2[o * COUT + k];
    float sb  = f_bias[o] * lrelu(s);
    float fwb = (6.0f - fminf(fmaxf(b_mask[o], 0.f), 6.f)) * (1.0f / 6.0f);
    nbias[g] = (1.0f - fwb) * ori_bias[o] + fwb * sb;
}

// ---------- global ori_weight stats (stage 1) ----------
__global__ __launch_bounds__(256) void k_ori_partial(const float* __restrict__ ow, float* __restrict__ part) {
    __shared__ float sh[512];
    float s = 0.f, s2 = 0.f;
    for (int i = blockIdx.x * 256 + threadIdx.x; i < NW; i += 64 * 256) {
        float w = ow[i]; s += w; s2 += w * w;
    }
    block_reduce2(s, s2, sh);
    if (threadIdx.x == 0) { part[blockIdx.x * 2] = s; part[blockIdx.x * 2 + 1] = s2; }
}

__device__ __forceinline__ float compute_w1(int e, int b, const float* ow, const float* wm,
                                            const float* fwgt, const float* style,
                                            float& fw_out, float& ori_out) {
    int o = e / 2304, r = e % 2304, i = r / 9, t = r % 9;
    float fw  = (6.0f - fminf(fmaxf(wm[o * CIN + i], 0.f), 6.f)) * (1.0f / 6.0f);
    float ori = ow[e];
    float sc  = style[b * STY + i * 9 + t];
    float sh  = style[b * STY + 2304 + i * 9 + t];
    float sw  = fwgt[e] * (sc + 1.0f) + sh;
    fw_out = fw; ori_out = ori;
    return (1.0f - fw) * ori + fw * sw;
}

// ---------- per-sample w1 stats (stage 1) ----------
__global__ __launch_bounds__(256) void k_w1_partial(const float* __restrict__ ow, const float* __restrict__ wm,
                                                    const float* __restrict__ fwgt, const float* __restrict__ style,
                                                    float* __restrict__ part) {
    __shared__ float sh[512];
    int b = blockIdx.y;
    float s = 0.f, s2 = 0.f;
    for (int e = blockIdx.x * 256 + threadIdx.x; e < NW; e += 64 * 256) {
        float fw, ori;
        float w1 = compute_w1(e, b, ow, wm, fwgt, style, fw, ori);
        s += w1; s2 += w1 * w1;
    }
    block_reduce2(s, s2, sh);
    if (threadIdx.x == 0) {
        part[(b * 64 + blockIdx.x) * 2]     = s;
        part[(b * 64 + blockIdx.x) * 2 + 1] = s2;
    }
}

// ---------- finalize stats: stats[0]=mean_ori stats[1]=std_ori stats[2+b]=mean stats[18+b]=std ----------
__global__ __launch_bounds__(256) void k_finalize(const float* __restrict__ opart, const float* __restrict__ spart,
                                                  float* __restrict__ stats) {
    __shared__ float sh[512];
    const float N = (float)NW;
    int tid = threadIdx.x;
    float s = 0.f, s2 = 0.f;
    if (tid < 64) { s = opart[tid * 2]; s2 = opart[tid * 2 + 1]; }
    block_reduce2(s, s2, sh);
    if (tid == 0) {
        stats[0] = s / N;
        stats[1] = sqrtf((s2 - s * s / N) / (N - 1.0f));
    }
    for (int b = 0; b < B_; ++b) {
        __syncthreads();
        float a = 0.f, a2 = 0.f;
        if (tid < 64) { a = spart[(b * 64 + tid) * 2]; a2 = spart[(b * 64 + tid) * 2 + 1]; }
        block_reduce2(a, a2, sh);
        if (tid == 0) {
            stats[2 + b]  = a / N;
            stats[18 + b] = sqrtf((a2 - a * a / N) / (N - 1.0f));
        }
    }
}

// ---------- new_weight -> bf16 [b][cout][tap][ci] ----------
__global__ __launch_bounds__(256) void k_new_weight(const float* __restrict__ ow, const float* __restrict__ wm,
                                                    const float* __restrict__ fwgt, const float* __restrict__ style,
                                                    const float* __restrict__ stats, __bf16* __restrict__ wbf) {
    int g = blockIdx.x * 256 + threadIdx.x;           // 16*589824
    int b = g / NW, e = g % NW;
    float fw, ori;
    float w1 = compute_w1(e, b, ow, wm, fwgt, style, fw, ori);
    float w2 = (w1 - stats[2 + b]) / stats[18 + b] * stats[1] + stats[0];
    float nw = (1.0f - fw) * ori + fw * w2;
    int o = e / 2304, r = e % 2304, i = r / 9, t = r % 9;
    wbf[((b * COUT + o) * 9 + t) * CIN + i] = (__bf16)nw;
}

// ---------- conv: 9 shifted GEMMs via v_wmma_f32_16x16x32_bf16 ----------
// grid: x = row (64), y = cout tile of 64 (4), z = b (16); block = 256 = 8 waves
// wave w owns cout subtile (w>>1) and two pixel subtiles of the (w&1) half-row:
//   A fragment loaded once per tap feeds two independent WMMA accumulation chains.
__global__ __launch_bounds__(256) void k_conv(const __bf16* __restrict__ yt, const __bf16* __restrict__ wbf,
                                              const float* __restrict__ nbias, float* __restrict__ out) {
    __shared__ alignas(32) __bf16 xl[3][66][32];      // [row][col(-1..64)][ci]        12,672 B
    __shared__ alignas(32) __bf16 wl[64][9][32];      // [cout][tap][ci]               36,864 B

    const int tid    = threadIdx.x;
    const int wave   = tid >> 5;
    const int lane   = tid & 31;
    const int lane16 = lane & 15;
    const bool hi    = lane >= 16;

    const int r     = blockIdx.x;
    const int o_blk = blockIdx.y * 64;
    const int b     = blockIdx.z;

    const int cout_sub = wave >> 1;                   // 0..3 -> 16 couts each
    const int pix_pair = wave & 1;                    // 0..1 -> 32 pixels each
    const int mrow     = cout_sub * 16 + lane16;      // A-matrix row owned by this lane
    const int n0       = pix_pair * 32 + lane16;      // pixel column, tile 0
    const int n1       = n0 + 16;                     // pixel column, tile 1
    const int arun     = hi ? 2 : 0;                  // A K-run start (v8bf units)
    const int bhalf    = hi ? 16 : 0;                 // B lanes>=16 hold K=16..31

    v8f acc0 = {0.f, 0.f, 0.f, 0.f, 0.f, 0.f, 0.f, 0.f};
    v8f acc1 = {0.f, 0.f, 0.f, 0.f, 0.f, 0.f, 0.f, 0.f};

    for (int c0 = 0; c0 < CIN; c0 += 32) {
        __syncthreads();
        // stage x: 3 rows x 66 cols x 32 ch as 16B chunks; interior chunks async, halo zero-filled
        for (int idx = tid; idx < 3 * 66 * 4; idx += 256) {
            int q = idx & 3, rc = idx >> 2;
            int col = rc % 66, row = rc / 66;
            int gr = r + row - 1, gc = col - 1;
            __bf16* l = &xl[row][col][q * 8];
            if (gr >= 0 && gr < H_ && gc >= 0 && gc < W_) {
                async_copy_b128(&yt[((b * H_ + gr) * W_ + gc) * CIN + c0 + q * 8], l);
            } else {
                v8bf z = {};
                *(v8bf*)l = z;
            }
        }
        // stage w: 64 cout x 9 taps x 32 ch as 16B chunks (always in-bounds -> pure async)
        for (int idx = tid; idx < 64 * 9 * 4; idx += 256) {
            int q = idx & 3, rt = idx >> 2;
            int t = rt % 9, oo = rt / 9;
            async_copy_b128(&wbf[((b * COUT + o_blk + oo) * 9 + t) * CIN + c0 + q * 8],
                            &wl[oo][t][q * 8]);
        }
        wait_async_lds();
        __syncthreads();

        #pragma unroll
        for (int ky = 0; ky < 3; ++ky) {
            #pragma unroll
            for (int kx = 0; kx < 3; ++kx) {
                const int t = ky * 3 + kx;
                // A fragment: 16 contiguous K values (K=16h..16h+15), matching B
                const v8bf* pw = reinterpret_cast<const v8bf*>(&wl[mrow][t][0]);
                v8bf a0 = pw[arun];
                v8bf a1 = pw[arun + 1];
                v16bf av;
                #pragma unroll
                for (int q = 0; q < 8; ++q) { av[q] = a0[q]; av[q + 8] = a1[q]; }
                // Two B fragments: 16 contiguous K values (channels) for two pixel columns
                v16bf bv0 = *reinterpret_cast<const v16bf*>(&xl[ky][n0 + kx][bhalf]);
                v16bf bv1 = *reinterpret_cast<const v16bf*>(&xl[ky][n1 + kx][bhalf]);
                acc0 = __builtin_amdgcn_wmma_f32_16x16x32_bf16(
                    false, av, false, bv0, (short)0, acc0, false, false);
                acc1 = __builtin_amdgcn_wmma_f32_16x16x32_bf16(
                    false, av, false, bv1, (short)0, acc1, false, false);
            }
        }
    }

    // store: VGPR j -> M = j + (hi?8:0), N = lane16 (two pixel tiles)
    #pragma unroll
    for (int j = 0; j < 8; ++j) {
        int o = o_blk + cout_sub * 16 + j + (hi ? 8 : 0);
        float bb = nbias[b * COUT + o];
        float* row = &out[((b * COUT + o) * H_ + r) * W_];
        row[n0] = acc0[j] + bb;
        row[n1] = acc1[j] + bb;
    }
}

extern "C" void kernel_launch(void* const* d_in, const int* in_sizes, int n_in,
                              void* d_out, int out_size, void* d_ws, size_t ws_size,
                              hipStream_t stream) {
    const float* y        = (const float*)d_in[0];
    const float* finger   = (const float*)d_in[1];
    const float* ori_w    = (const float*)d_in[2];
    const float* w_mask   = (const float*)d_in[3];
    const float* f_weight = (const float*)d_in[4];
    const float* style_w1 = (const float*)d_in[5];
    const float* style_w2 = (const float*)d_in[6];
    const float* ori_bias = (const float*)d_in[7];
    const float* b_mask   = (const float*)d_in[8];
    const float* f_bias   = (const float*)d_in[9];
    const float* bias_w1  = (const float*)d_in[10];
    const float* bias_w2  = (const float*)d_in[11];
    float* out = (float*)d_out;

    char* ws = (char*)d_ws;
    __bf16* yt    = (__bf16*)(ws);                       // 33,554,432 B
    __bf16* wbf   = (__bf16*)(ws + 33554432);            // 18,874,368 B
    float*  style = (float*)(ws + 52428800);             //    294,912 B
    float*  hbuf  = (float*)(ws + 52723712);             //      8,192 B
    float*  t1    = (float*)(ws + 52731904);             //     16,384 B
    float*  nbias = (float*)(ws + 52748288);             //     16,384 B
    float*  opart = (float*)(ws + 52764672);             //        512 B
    float*  spart = (float*)(ws + 52765184);             //      8,192 B
    float*  stats = (float*)(ws + 52773376);             //        160 B

    k_convert_y<<<65536, 256, 0, stream>>>(y, yt);
    k_h<<<8, 256, 0, stream>>>(finger, style_w1, hbuf);
    k_style<<<288, 256, 0, stream>>>(hbuf, style_w2, style);
    k_bias1<<<16, 256, 0, stream>>>(finger, bias_w1, t1);
    k_bias2<<<16, 256, 0, stream>>>(t1, bias_w2, f_bias, b_mask, ori_bias, nbias);
    k_ori_partial<<<64, 256, 0, stream>>>(ori_w, opart);
    k_w1_partial<<<dim3(64, 16), 256, 0, stream>>>(ori_w, w_mask, f_weight, style, spart);
    k_finalize<<<1, 256, 0, stream>>>(opart, spart, stats);
    k_new_weight<<<36864, 256, 0, stream>>>(ori_w, w_mask, f_weight, style, stats, wbf);
    k_conv<<<dim3(64, 4, 16), 256, 0, stream>>>(yt, wbf, nbias, out);
}
